// CombinedPairwiseCacheLoss_88656714924228
// MI455X (gfx1250) — matrix operand order repaired and checked by reading, not compiled, not run on hardware
//
#include <hip/hip_runtime.h>

#define NROWS  1024
#define DIM    1024
#define MCACHE 10000
#define MPAD   10112          // 158 * 64
#define NCB    158            // column blocks of 64
#define GAMMA  30.0f
#define MARGIN 0.4f

typedef __attribute__((ext_vector_type(16))) __bf16 v16bf;
typedef __attribute__((ext_vector_type(8)))  float  v8f;
typedef __attribute__((ext_vector_type(4)))  int    v4i;

typedef __attribute__((address_space(1))) v4i* gv4i_ptr;   // global (AS1), non-const
typedef __attribute__((address_space(3))) v4i* lv4i_ptr;   // LDS (AS3)

// Async global->LDS copy path (CDNA5 GLOBAL_LOAD_ASYNC_TO_LDS_B128, ASYNCcnt).
#if defined(__gfx1250__) && __has_builtin(__builtin_amdgcn_global_load_async_to_lds_b128)
#define ASYNC_LDS 1
#else
#define ASYNC_LDS 0
#endif

// ---------- helpers ----------
__device__ __forceinline__ unsigned short f2bf(float f) {
  unsigned int u = __float_as_uint(f);
  u = (u + 0x7FFFu + ((u >> 16) & 1u)) >> 16;   // round-to-nearest-even
  return (unsigned short)u;
}

// online logsumexp: push one element x into (m, s) where lse = m + log(s)
__device__ __forceinline__ void lse_push(float& m, float& s, float x) {
  if (s <= 0.0f) { m = x; s = 1.0f; return; }
  if (x > m) { s = s * __expf(m - x) + 1.0f; m = x; }
  else       { s += __expf(x - m); }
}

// combine two (max, sum) logsumexp states
__device__ __forceinline__ void lse_combine(float& m, float& s, float m2, float s2) {
  if (s2 <= 0.0f) return;
  if (s <= 0.0f) { m = m2; s = s2; return; }
  if (m2 > m) { s = s * __expf(m - m2) + s2; m = m2; }
  else        { s += s2 * __expf(m2 - m); }
}

__device__ __forceinline__ void wait_async_copies() {
#if ASYNC_LDS
#if __has_builtin(__builtin_amdgcn_s_wait_asynccnt)
  __builtin_amdgcn_s_wait_asynccnt(0);
#else
  asm volatile("s_wait_asynccnt 0x0" ::: "memory");
#endif
#endif
}

// stage one 128x32 A tile + 64x32 B tile (bf16) into the given LDS buffers
__device__ __forceinline__ void stage_tiles(const unsigned short* __restrict__ cacheBf,
                                            uint4* __restrict__ ldsA,
                                            uint4* __restrict__ ldsB,
                                            int tid, int rowBase, int colBase, int kb) {
#pragma unroll
  for (int t2 = 0; t2 < 2; ++t2) {
    const int u = tid * 2 + t2;
    const int r = u >> 2, seg = u & 3;
    const unsigned short* g = cacheBf + (size_t)(rowBase + r) * DIM + kb + seg * 8;
#if ASYNC_LDS
    __builtin_amdgcn_global_load_async_to_lds_b128(
        (gv4i_ptr)(uintptr_t)g,
        (lv4i_ptr)&ldsA[u], 0, 0);
#else
    ldsA[u] = *(const uint4*)g;
#endif
  }
  {
    const int r = tid >> 2, seg = tid & 3;
    const unsigned short* g = cacheBf + (size_t)(colBase + r) * DIM + kb + seg * 8;
#if ASYNC_LDS
    __builtin_amdgcn_global_load_async_to_lds_b128(
        (gv4i_ptr)(uintptr_t)g,
        (lv4i_ptr)&ldsB[tid], 0, 0);
#else
    ldsB[tid] = *(const uint4*)g;
    if (kb + 32 < DIM)  // hint next chunk (global_prefetch_b8) on sync path
      __builtin_prefetch(g + 32, 0, 1);
#endif
  }
}

// ---------- kernel 1: L2-normalize embedding rows -> bf16 cache rows 0..1023 ----------
__global__ void __launch_bounds__(256) norm_bf16_kernel(const float* __restrict__ emb,
                                                        unsigned short* __restrict__ dst) {
  __shared__ float red[256];
  const int row = blockIdx.x;
  const int t   = threadIdx.x;
  const float4 v = ((const float4*)(emb + (size_t)row * DIM))[t];
  red[t] = v.x * v.x + v.y * v.y + v.z * v.z + v.w * v.w;
  __syncthreads();
  for (int o = 128; o > 0; o >>= 1) {
    if (t < o) red[t] += red[t + o];
    __syncthreads();
  }
  const float inv = 1.0f / sqrtf(red[0]);
  ushort4 o4;
  o4.x = f2bf(v.x * inv); o4.y = f2bf(v.y * inv);
  o4.z = f2bf(v.z * inv); o4.w = f2bf(v.w * inv);
  ((ushort4*)(dst + (size_t)row * DIM))[t] = o4;
}

// ---------- kernel 2: convert old cache rows -> bf16 rows 1024..9999; zero rows 10000..10111 ----------
__global__ void __launch_bounds__(256) cache_convert_kernel(const float* __restrict__ oldf,
                                                            unsigned short* __restrict__ dst) {
  const size_t e  = ((size_t)blockIdx.x * 256 + threadIdx.x) * 4;
  const int row   = NROWS + (int)(e >> 10);
  const int col   = (int)(e & 1023);
  ushort4 o4;
  if (row < MCACHE) {
    const float4 v = *(const float4*)(oldf + (size_t)(row - NROWS) * DIM + col);
    o4.x = f2bf(v.x); o4.y = f2bf(v.y); o4.z = f2bf(v.z); o4.w = f2bf(v.w);
  } else {
    o4.x = 0; o4.y = 0; o4.z = 0; o4.w = 0;   // padded columns
  }
  *(ushort4*)(dst + (size_t)row * DIM + col) = o4;
}

// ---------- kernel 3: WMMA GEMM (emb @ cache^T) with fused circle-loss logsumexp epilogue ----------
__global__ void __launch_bounds__(256) gemm_loss_kernel(const unsigned short* __restrict__ cacheBf,
                                                        const int* __restrict__ targets,
                                                        const int* __restrict__ oldLabels,
                                                        float4* __restrict__ partials) {
  __shared__ uint4 ldsA[2][512];   // double-buffered 128 x 32 bf16
  __shared__ uint4 ldsB[2][256];   // double-buffered  64 x 32 bf16
  const int tid  = threadIdx.x;
  const int lane = tid & 31;
  const int wm   = tid >> 5;          // wave id -> row strip (8 waves x 16 rows = 128)
  const int h    = lane >> 4;         // half-wave
  const int lm   = lane & 15;
  const int rowBase = blockIdx.y * 128;
  const int colBase = blockIdx.x * 64;

  v8f acc[4];
  const v8f vzero = {0.f, 0.f, 0.f, 0.f, 0.f, 0.f, 0.f, 0.f};
#pragma unroll
  for (int n = 0; n < 4; ++n) acc[n] = vzero;

  constexpr int KCH = DIM / 32;        // 32 K-chunks
  stage_tiles(cacheBf, ldsA[0], ldsB[0], tid, rowBase, colBase, 0);

  for (int i = 0; i < KCH; ++i) {
    const int cur = i & 1;
    wait_async_copies();               // stage(i) complete (this wave's async ops)
    __syncthreads();                   // all waves' copies visible / all reads of cur^1 done
    if (i + 1 < KCH)                   // overlap copy of chunk i+1 with WMMAs on chunk i
      stage_tiles(cacheBf, ldsA[cur ^ 1], ldsB[cur ^ 1], tid, rowBase, colBase, (i + 1) * 32);

    // A fragment (16x32, ISA 7.12.2 16-bit A layout: K = {8h..8h+7, 8h+16..8h+23})
    union { uint4 q[2]; v16bf v; } af;
    af.q[0] = ldsA[cur][(wm * 16 + lm) * 4 + h];
    af.q[1] = ldsA[cur][(wm * 16 + lm) * 4 + h + 2];

#pragma unroll
    for (int n = 0; n < 4; ++n) {
      // B fragment (32x16: lane = column (cache row), K = {8h..8h+7, 8h+16..8h+23})
      union { uint4 q[2]; v16bf v; } bf_;
      bf_.q[0] = ldsB[cur][(n * 16 + lm) * 4 + h];
      bf_.q[1] = ldsB[cur][(n * 16 + lm) * 4 + h + 2];
      acc[n] = __builtin_amdgcn_wmma_f32_16x16x32_bf16(
          false, af.v, false, bf_.v, (short)0, acc[n], false, false);
    }
  }

  // ---- fused circle-loss epilogue ----
  // C layout: lane element r -> (row m = 8h + r, col n = lm)
  int ti[8];
#pragma unroll
  for (int r = 0; r < 8; ++r) ti[r] = targets[rowBase + wm * 16 + h * 8 + r];

  float mp[8], sp[8], mn[8], sn[8];
#pragma unroll
  for (int r = 0; r < 8; ++r) { mp[r] = -INFINITY; sp[r] = 0.f; mn[r] = -INFINITY; sn[r] = 0.f; }

#pragma unroll
  for (int n = 0; n < 4; ++n) {
    const int j = colBase + n * 16 + lm;
    if (j < MCACHE) {
      const int lab = (j < NROWS) ? targets[j] : oldLabels[j - NROWS];
#pragma unroll
      for (int r = 0; r < 8; ++r) {
        const float s  = acc[n][r];
        const int   gi = rowBase + wm * 16 + h * 8 + r;
        if (lab == ti[r]) {
          if (j != gi) {   // self-match removed (reference eye subtraction)
            const float ap = fmaxf(1.0f + MARGIN - s, 0.0f);
            lse_push(mp[r], sp[r], -GAMMA * ap * (s - (1.0f - MARGIN)));
          }
        } else {
          const float an = fmaxf(s + MARGIN, 0.0f);
          lse_push(mn[r], sn[r], GAMMA * an * (s - MARGIN));
        }
      }
    }
  }

  // reduce across the 16 column-lanes of each half-wave
#pragma unroll
  for (int r = 0; r < 8; ++r) {
#pragma unroll
    for (int mask = 1; mask <= 8; mask <<= 1) {
      float om = __shfl_xor(mp[r], mask, 32);
      float os = __shfl_xor(sp[r], mask, 32);
      lse_combine(mp[r], sp[r], om, os);
      om = __shfl_xor(mn[r], mask, 32);
      os = __shfl_xor(sn[r], mask, 32);
      lse_combine(mn[r], sn[r], om, os);
    }
  }

  if (lm == 0) {
#pragma unroll
    for (int r = 0; r < 8; ++r) {
      const int grow = rowBase + wm * 16 + h * 8 + r;
      partials[(size_t)grow * NCB + blockIdx.x] = make_float4(mp[r], sp[r], mn[r], sn[r]);
    }
  }
}

// ---------- kernel 4: fold column-block partials, softplus, mean ----------
__global__ void __launch_bounds__(1024) reduce_kernel(const float4* __restrict__ partials,
                                                      float* __restrict__ out) {
  __shared__ float red[1024];
  const int i = threadIdx.x;   // one row per thread
  float mp = -INFINITY, sp = 0.f, mn = -INFINITY, sn = 0.f;
  for (int cb = 0; cb < NCB; ++cb) {
    const float4 p = partials[(size_t)i * NCB + cb];
    lse_combine(mp, sp, p.x, p.y);
    lse_combine(mn, sn, p.z, p.w);
  }
  const float lp  = (sp > 0.f) ? mp + __logf(sp) : -INFINITY;
  const float ln  = (sn > 0.f) ? mn + __logf(sn) : -INFINITY;
  const float tot = lp + ln;
  float v;
  if (tot > 20.f)       v = tot;                 // softplus(x) ~ x
  else if (tot < -20.f) v = __expf(tot);         // softplus(x) ~ e^x (0 at -inf)
  else                  v = log1pf(__expf(tot));
  red[i] = v;
  __syncthreads();
  for (int o = 512; o > 0; o >>= 1) {
    if (i < o) red[i] += red[i + o];
    __syncthreads();
  }
  if (i == 0) out[0] = red[0] * (1.0f / (float)NROWS);
}

// ---------- launcher ----------
extern "C" void kernel_launch(void* const* d_in, const int* in_sizes, int n_in,
                              void* d_out, int out_size, void* d_ws, size_t ws_size,
                              hipStream_t stream) {
  const float* embedding = (const float*)d_in[0];
  const float* old_cache = (const float*)d_in[1];
  const int*   targets   = (const int*)d_in[2];
  const int*   oldLab    = (const int*)d_in[3];
  float*       out       = (float*)d_out;

  // workspace layout: [ bf16 cache (MPAD x DIM) | float4 partials (NROWS x NCB) ]
  unsigned short* cacheBf  = (unsigned short*)d_ws;
  float4*         partials = (float4*)((char*)d_ws + (size_t)MPAD * DIM * sizeof(unsigned short));

  norm_bf16_kernel<<<NROWS, 256, 0, stream>>>(embedding, cacheBf);
  cache_convert_kernel<<<(MPAD - NROWS), 256, 0, stream>>>(old_cache, cacheBf);
  dim3 grid(NCB, NROWS / 128);
  gemm_loss_kernel<<<grid, 256, 0, stream>>>(cacheBf, targets, oldLab, partials);
  reduce_kernel<<<1, 1024, 0, stream>>>(partials, out);
}
